// SwiGLUFFN_75368086110865
// MI455X (gfx1250) — 8-bit matrix operand order repaired and checked by reading, compiled, not run on hardware
//
#include <hip/hip_runtime.h>
#include <hip/hip_bf16.h>
#include <math.h>


typedef __attribute__((ext_vector_type(8))) int v8i;

#define RED_THREADS 256


__global__ void absmean_partial(const float* __restrict__ w, long n,
                                float* __restrict__ partial) {
  __shared__ float red[RED_THREADS];
  int tid = threadIdx.x;
  float s = 0.f;
  for (long i = (long)blockIdx.x * blockDim.x + tid; i < n;
       i += (long)gridDim.x * blockDim.x)
    s += fabsf(w[i]);
  red[tid] = s;
  __syncthreads();
  for (int o = RED_THREADS / 2; o > 0; o >>= 1) {
    if (tid < o) red[tid] += red[tid + o];
    __syncthreads();
  }
  if (tid == 0) partial[blockIdx.x] = red[0];
}

__global__ void absmean_final(const float* __restrict__ partial, int nb,
                              float inv_n, float* __restrict__ out_mul) {
  __shared__ float red[RED_THREADS];
  int tid = threadIdx.x;
  float s = 0.f;
  for (int i = tid; i < nb; i += RED_THREADS) s += partial[i];
  red[tid] = s;
  __syncthreads();
  for (int o = RED_THREADS / 2; o > 0; o >>= 1) {
    if (tid < o) red[tid] += red[tid + o];
    __syncthreads();
  }
  if (tid == 0) out_mul[0] = fmaxf(red[0] * inv_n, 1e-5f);
}

__global__ void quant_weight(const float* __restrict__ w, long n,
                             const float* __restrict__ mul,
                             signed char* __restrict__ wq) {
  float scale = 1.f / mul[0];
  for (long i = (long)blockIdx.x * blockDim.x + threadIdx.x; i < n;
       i += (long)gridDim.x * blockDim.x) {
    float q = rintf(w[i] * scale);
    q = fminf(1.f, fmaxf(-1.f, q));
    wq[i] = (signed char)(int)q;
  }
}

__global__ void act_rms_quant(const float* __restrict__ x, int L,
                              signed char* __restrict__ q,
                              float* __restrict__ s_inv) {
  const int t = blockIdx.x;
  const float* row = x + (long)t * L;
  __shared__ float red[RED_THREADS];
  __shared__ float s_rn, s_sc;
  int tid = threadIdx.x;

  float ss = 0.f;
  for (int i = tid; i < L; i += RED_THREADS) {
    float v = row[i];
    ss += v * v;
  }
  red[tid] = ss;
  __syncthreads();
  for (int o = RED_THREADS / 2; o > 0; o >>= 1) {
    if (tid < o) red[tid] += red[tid + o];
    __syncthreads();
  }
  if (tid == 0) s_rn = rsqrtf(red[0] / (float)L + 1e-6f);
  __syncthreads();
  float rn = s_rn;

  float am = 0.f;
  for (int i = tid; i < L; i += RED_THREADS)
    am = fmaxf(am, fabsf(row[i] * rn));
  red[tid] = am;
  __syncthreads();
  for (int o = RED_THREADS / 2; o > 0; o >>= 1) {
    if (tid < o) red[tid] = fmaxf(red[tid], red[tid + o]);
    __syncthreads();
  }
  if (tid == 0) {
    float amax = fmaxf(red[0], 1e-5f);
    s_sc = 127.f / amax;
    s_inv[t] = amax / 127.f;
  }
  __syncthreads();
  float sc = s_sc;

  signed char* qrow = q + (long)t * L;
  for (int i = tid; i < L; i += RED_THREADS) {
    float v = rintf(row[i] * rn * sc);
    v = fminf(127.f, fmaxf(-128.f, v));
    qrow[i] = (signed char)(int)v;
  }
}

__device__ __forceinline__ v8i load_a_frag(const signed char* __restrict__ base,
                                           int ldk, int row, int k, int lane) {
  const signed char* p =
      base + (long)(row + (lane & 15)) * ldk + k + ((lane >> 4) << 4);
  int4 c0 = *reinterpret_cast<const int4*>(p);
  int4 c1 = *reinterpret_cast<const int4*>(p + 32);
  v8i f = {c0.x, c0.y, c0.z, c0.w, c1.x, c1.y, c1.z, c1.w};
  return f;
}

__device__ __forceinline__ v8i load_b_frag(const signed char* __restrict__ base,
                                           int ldk, int col, int k, int lane) {
  const signed char* p =
      base + (long)(col + (lane & 15)) * ldk + k + ((lane >> 4) << 4);
  int4 c0 = *reinterpret_cast<const int4*>(p);
  int4 c1 = *reinterpret_cast<const int4*>(p + 32);
  v8i f = {c0.x, c0.y, c0.z, c0.w, c1.x, c1.y, c1.z, c1.w};
  return f;
}

__device__ __forceinline__ v8i wmma_iu8(v8i a, v8i b, v8i c) {
  return __builtin_amdgcn_wmma_i32_16x16x64_iu8(true, a, true, b, c, false,
                                                false);
}

__device__ __forceinline__ void store_h_tile(float* __restrict__ h, int H,
                                             int m, int n, int lane, v8i g,
                                             v8i u, float gm, float um,
                                             const float* __restrict__ sx_inv) {
  int col = n + (lane & 15);
  int rofs = (lane >> 4) << 3;
#pragma unroll
  for (int i = 0; i < 8; ++i) {
    int row = m + rofs + i;
    float sx = sx_inv[row];
    float gf = (float)g[i] * sx * gm;
    float uf = (float)u[i] * sx * um;
    float sg = gf / (1.f + __expf(-gf));
    h[(long)row * H + col] = sg * uf;
  }
}

__device__ __forceinline__ void store_o_tile(float* __restrict__ o, int N,
                                             int m, int n, int lane, v8i acc,
                                             float dm,
                                             const float* __restrict__ s_inv) {
  int col = n + (lane & 15);
  int rofs = (lane >> 4) << 3;
#pragma unroll
  for (int i = 0; i < 8; ++i) {
    int row = m + rofs + i;
    o[(long)row * N + col] = (float)acc[i] * s_inv[row] * dm;
  }
}

__global__ __launch_bounds__(256) void bit_gemm_gate_up(
    const signed char* __restrict__ xq, const float* __restrict__ sx_inv,
    const signed char* __restrict__ wgq, const signed char* __restrict__ wuq,
    const float* __restrict__ muls,
    float* __restrict__ h, int K, int H) {
  const int lane = threadIdx.x & 31;
  const int wave = threadIdx.x >> 5;
  const int m_base = blockIdx.x * 128 + (wave >> 1) * 32;
  const int n_base = blockIdx.y * 64 + (wave & 1) * 32;

  v8i z = {0, 0, 0, 0, 0, 0, 0, 0};
  v8i g00 = z, g01 = z, g10 = z, g11 = z;
  v8i u00 = z, u01 = z, u10 = z, u11 = z;

  for (int k = 0; k < K; k += 64) {
    v8i a0 = load_a_frag(xq, K, m_base, k, lane);
    v8i a1 = load_a_frag(xq, K, m_base + 16, k, lane);
    v8i bg0 = load_b_frag(wgq, K, n_base, k, lane);
    v8i bg1 = load_b_frag(wgq, K, n_base + 16, k, lane);
    v8i bu0 = load_b_frag(wuq, K, n_base, k, lane);
    v8i bu1 = load_b_frag(wuq, K, n_base + 16, k, lane);
    if (k + 64 < K) {
      __builtin_prefetch(wgq + (long)(n_base + (lane & 15)) * K + k + 64, 0, 1);
      __builtin_prefetch(wuq + (long)(n_base + (lane & 15)) * K + k + 64, 0, 1);
    }
    g00 = wmma_iu8(a0, bg0, g00);
    g01 = wmma_iu8(a0, bg1, g01);
    g10 = wmma_iu8(a1, bg0, g10);
    g11 = wmma_iu8(a1, bg1, g11);
    u00 = wmma_iu8(a0, bu0, u00);
    u01 = wmma_iu8(a0, bu1, u01);
    u10 = wmma_iu8(a1, bu0, u10);
    u11 = wmma_iu8(a1, bu1, u11);
  }

  float gm = muls[0], um = muls[1];
  store_h_tile(h, H, m_base, n_base, lane, g00, u00, gm, um, sx_inv);
  store_h_tile(h, H, m_base, n_base + 16, lane, g01, u01, gm, um, sx_inv);
  store_h_tile(h, H, m_base + 16, n_base, lane, g10, u10, gm, um, sx_inv);
  store_h_tile(h, H, m_base + 16, n_base + 16, lane, g11, u11, gm, um, sx_inv);
}

__global__ __launch_bounds__(256) void bit_gemm_down(
    const signed char* __restrict__ hq, const float* __restrict__ sh_inv,
    const signed char* __restrict__ wdq,
    const float* __restrict__ muls,
    float* __restrict__ out, int K, int N) {
  const int lane = threadIdx.x & 31;
  const int wave = threadIdx.x >> 5;
  const int m_base = blockIdx.x * 128 + (wave >> 1) * 32;
  const int n_base = blockIdx.y * 64 + (wave & 1) * 32;

  v8i z = {0, 0, 0, 0, 0, 0, 0, 0};
  v8i c00 = z, c01 = z, c10 = z, c11 = z;

  for (int k = 0; k < K; k += 64) {
    v8i a0 = load_a_frag(hq, K, m_base, k, lane);
    v8i a1 = load_a_frag(hq, K, m_base + 16, k, lane);
    v8i b0 = load_b_frag(wdq, K, n_base, k, lane);
    v8i b1 = load_b_frag(wdq, K, n_base + 16, k, lane);
    if (k + 64 < K)
      __builtin_prefetch(wdq + (long)(n_base + (lane & 15)) * K + k + 64, 0, 1);
    c00 = wmma_iu8(a0, b0, c00);
    c01 = wmma_iu8(a0, b1, c01);
    c10 = wmma_iu8(a1, b0, c10);
    c11 = wmma_iu8(a1, b1, c11);
  }

  float dm = muls[2];
  store_o_tile(out, N, m_base, n_base, lane, c00, dm, sh_inv);
  store_o_tile(out, N, m_base, n_base + 16, lane, c01, dm, sh_inv);
  store_o_tile(out, N, m_base + 16, n_base, lane, c10, dm, sh_inv);
  store_o_tile(out, N, m_base + 16, n_base + 16, lane, c11, dm, sh_inv);
}

extern "C" void kernel_launch(void* const* d_in, const int* in_sizes, int n_in,
                              void* d_out, int out_size, void* d_ws,
                              size_t ws_size, hipStream_t stream) {
  const float* x = (const float*)d_in[0];
  const float* w_gate = (const float*)d_in[1];
  const float* w_up = (const float*)d_in[2];
  const float* w_down = (const float*)d_in[3];
  float* out = (float*)d_out;

  const int D = 2048;
  const int H = 8192;
  const int T = in_sizes[0] / D;
  const long nW = (long)H * D;
  const int NPART = 1024;

  char* wsb = (char*)d_ws;
  size_t off = 0;
  auto take = [&](size_t bytes) -> char* {
    char* p = wsb + off;
    off = (off + bytes + 255) & ~(size_t)255;
    return p;
  };
  signed char* xq = (signed char*)take((size_t)T * D);
  float* sx_inv = (float*)take((size_t)T * sizeof(float));
  signed char* wgq = (signed char*)take((size_t)nW);
  signed char* wuq = (signed char*)take((size_t)nW);
  signed char* wdq = (signed char*)take((size_t)nW);
  float* hbuf = (float*)take((size_t)T * H * sizeof(float));
  signed char* hq = (signed char*)take((size_t)T * H);
  float* sh_inv = (float*)take((size_t)T * sizeof(float));
  float* partials = (float*)take((size_t)NPART * sizeof(float));
  float* muls = (float*)take(4 * sizeof(float));

  float inv_nW = 1.f / (float)nW;
  absmean_partial<<<NPART, RED_THREADS, 0, stream>>>(w_gate, nW, partials);
  absmean_final<<<1, RED_THREADS, 0, stream>>>(partials, NPART, inv_nW, muls + 0);
  absmean_partial<<<NPART, RED_THREADS, 0, stream>>>(w_up, nW, partials);
  absmean_final<<<1, RED_THREADS, 0, stream>>>(partials, NPART, inv_nW, muls + 1);
  absmean_partial<<<NPART, RED_THREADS, 0, stream>>>(w_down, nW, partials);
  absmean_final<<<1, RED_THREADS, 0, stream>>>(partials, NPART, inv_nW, muls + 2);

  quant_weight<<<4096, RED_THREADS, 0, stream>>>(w_gate, nW, muls + 0, wgq);
  quant_weight<<<4096, RED_THREADS, 0, stream>>>(w_up, nW, muls + 1, wuq);
  quant_weight<<<4096, RED_THREADS, 0, stream>>>(w_down, nW, muls + 2, wdq);

  act_rms_quant<<<T, RED_THREADS, 0, stream>>>(x, D, xq, sx_inv);

  bit_gemm_gate_up<<<dim3(T / 128, H / 64), 256, 0, stream>>>(
      xq, sx_inv, wgq, wuq, muls, hbuf, D, H);

  act_rms_quant<<<T, RED_THREADS, 0, stream>>>(hbuf, H, hq, sh_inv);

  bit_gemm_down<<<dim3(T / 128, D / 64), 256, 0, stream>>>(
      hq, sh_inv, wdq, muls, out, H, D);
}
